// EquiGraspSO3DeformableAttn2_3410204033231
// MI455X (gfx1250) — matrix operand order repaired and checked by reading, not compiled, not run on hardware
//
#include <hip/hip_runtime.h>

typedef __attribute__((ext_vector_type(16))) _Float16 v16h;
typedef __attribute__((ext_vector_type(8)))  _Float16 v8h;
typedef __attribute__((ext_vector_type(8)))  float    v8f;

#define FDIM   128
#define NCP    25
#define HPLANE 128
#define SROW   136   // padded LDS row stride (halves): 272B row -> bank-friendly, 16B aligned

// ---- WMMA fragment loaders -------------------------------------------------
// A: 16x32 f16 tile, row-major in LDS (stride SROW halves).
// ISA layout: lane<16 holds row (lane&15), K = {kb..kb+7} U {kb+16..kb+23}, kb = (lane>>4)*8.
__device__ __forceinline__ v16h load_a_frag(const _Float16* base, int k0, int lane) {
  int r  = lane & 15;
  int kb = (lane >> 4) << 3;
  const _Float16* p = base + r * SROW + k0 + kb;
  v8h lo = *(const v8h*)(p);        // ds_read_b128
  v8h hi = *(const v8h*)(p + 16);   // ds_read_b128
  v16h f;
#pragma unroll
  for (int i = 0; i < 8; ++i) { f[i] = lo[i]; f[i + 8] = hi[i]; }
  return f;
}

// B: 32x16 tile from a global K x ncols row-major f32 matrix, converted to f16.
// ISA layout: lane holds column (lane&15), K = {kb..kb+7} U {kb+16..kb+23}, kb = (lane>>4)*8.
// Columns >= ncols are zero-padded (used for W_w with N=25 padded to 32).
__device__ __forceinline__ v16h load_b_frag_g(const float* __restrict__ W, int ncols,
                                              int n0, int k0, int lane) {
  int n  = (lane & 15) + n0;
  int kb = (lane >> 4) << 3;
  v16h f;
  if (n < ncols) {
    const float* p = W + (size_t)(k0 + kb) * ncols + n;
#pragma unroll
    for (int j = 0; j < 16; ++j) f[j] = (_Float16)p[(size_t)(j + (j & 8)) * ncols];
  } else {
#pragma unroll
    for (int j = 0; j < 16; ++j) f[j] = (_Float16)0.f;
  }
  return f;
}

// ---- Bilinear plane sampler (grid_sample align_corners=True, border clamp) --
// plane: (C=128, H=128, W=128) f32 for one batch. Each lane accumulates 4
// channels: c = lane + 32*j. acc[j] += w * bilinear(plane[c], u->W axis, v->H axis).
__device__ __forceinline__ void sample_plane(const float* __restrict__ plane,
                                             float u, float v, float w,
                                             float acc[4], int lane) {
  float x = fminf(fmaxf(u, 0.f), 1.f) * 127.f;
  float y = fminf(fmaxf(v, 0.f), 1.f) * 127.f;
  int x0 = (int)x; if (x0 > 126) x0 = 126;
  int y0 = (int)y; if (y0 > 126) y0 = 126;
  float wx = x - (float)x0;
  float wy = y - (float)y0;
  float w00 = (1.f - wx) * (1.f - wy) * w;
  float w01 = wx * (1.f - wy) * w;
  float w10 = (1.f - wx) * wy * w;
  float w11 = wx * wy * w;
  const float* p0 = plane + (y0 * HPLANE + x0);
#pragma unroll
  for (int j = 0; j < 4; ++j) {
    const float* pc = p0 + (size_t)(lane + (j << 5)) * (HPLANE * HPLANE);
    float f00 = pc[0];
    float f01 = pc[1];
    float f10 = pc[HPLANE];
    float f11 = pc[HPLANE + 1];
    acc[j] = fmaf(w00, f00, fmaf(w01, f01, fmaf(w10, f10, fmaf(w11, f11, acc[j]))));
  }
}

__global__ __launch_bounds__(128)
void EquiGraspSO3DeformableAttn2_3410204033231_kernel(
    const float* __restrict__ qp,  const float* __restrict__ cxz,
    const float* __restrict__ cxy, const float* __restrict__ cyz,
    const float* __restrict__ cpg, const float* __restrict__ Wv,
    const float* __restrict__ bv,  const float* __restrict__ Ww,
    const float* __restrict__ bw,  const float* __restrict__ Wo,
    const float* __restrict__ bo,  float* __restrict__ out) {
  __shared__ float s_bv[FDIM];
  __shared__ float s_bo[FDIM];
  __shared__ float s_bw[32];
  __shared__ float s_cp[NCP * 3 + 5];
  __shared__ float s_posori[4][16 * 9];
  __shared__ _Float16 __attribute__((aligned(16))) s_feat[4][16 * SROW];
  __shared__ _Float16 __attribute__((aligned(16))) s_ws[4][16 * SROW];
  __shared__ float s_wts[4][16 * 32];
  __shared__ float s_sumw[4][16];

  const int tid  = threadIdx.x;
  const int lane = tid & 31;
  const int wave = tid >> 5;

  // ---- Block-wide staging of small constants ----
  for (int i = tid; i < FDIM; i += 128) { s_bv[i] = bv[i]; s_bo[i] = bo[i]; }
  if (tid < 32)      s_bw[tid] = (tid < NCP) ? bw[tid] : 0.f;
  if (tid < NCP * 3) s_cp[tid] = cpg[tid];

  const int pt0 = blockIdx.x * 64 + wave * 16;   // 4 waves * 16 points
  for (int i = lane; i < 16 * 9; i += 32)
    s_posori[wave][i] = qp[(size_t)pt0 * 9 + i];
  __syncthreads();

  const int b = pt0 >> 10;                       // 1024 points per batch
  const size_t pstride = (size_t)FDIM * HPLANE * HPLANE;
  const float* pxz = cxz + (size_t)b * pstride;
  const float* pxy = cxy + (size_t)b * pstride;
  const float* pyz = cyz + (size_t)b * pstride;

  _Float16* feat_h = s_feat[wave];
  _Float16* ws_h   = s_ws[wave];
  float*    wts    = s_wts[wave];
  float*    posori = s_posori[wave];

  // ---- Phase 1: feature = sum of 3 plane samples at query pos ----
#pragma unroll 1
  for (int m = 0; m < 16; ++m) {
    float px = posori[m * 9 + 0], py = posori[m * 9 + 1], pz = posori[m * 9 + 2];
    float acc[4] = {0.f, 0.f, 0.f, 0.f};
    sample_plane(pxz, px, pz, 1.f, acc, lane);
    sample_plane(pxy, px, py, 1.f, acc, lane);
    sample_plane(pyz, py, pz, 1.f, acc, lane);
#pragma unroll
    for (int j = 0; j < 4; ++j)
      feat_h[m * SROW + lane + (j << 5)] = (_Float16)acc[j];
  }
  asm volatile("s_wait_dscnt 0" ::: "memory");   // cross-lane LDS handoff

  // ---- Phase 2: weights = feat(16x128) @ W_w(128x25->32) + b_w via WMMA ----
  {
    v16h af[4];
#pragma unroll
    for (int k = 0; k < 4; ++k) af[k] = load_a_frag(feat_h, k * 32, lane);
    const int n = lane & 15, hif = lane >> 4;
#pragma unroll
    for (int t = 0; t < 2; ++t) {
      v8f c = {};
#pragma unroll
      for (int k = 0; k < 4; ++k) {
        v16h bf = load_b_frag_g(Ww, NCP, t * 16, k * 32, lane);
        c = __builtin_amdgcn_wmma_f32_16x16x32_f16(false, af[k], false, bf,
                                                   (short)0, c, false, false);
      }
      float bwg = s_bw[t * 16 + n];
#pragma unroll
      for (int r = 0; r < 8; ++r)
        wts[(r + 8 * hif) * 32 + t * 16 + n] = c[r] + bwg;   // D: M=r+8*hi, N=lane%16
    }
  }
  asm volatile("s_wait_dscnt 0" ::: "memory");
  if (lane < 16) {                               // sumw for the b_v term
    float ssum = 0.f;
    for (int g = 0; g < NCP; ++g) ssum += wts[lane * 32 + g];
    s_sumw[wave][lane] = ssum;
  }

  // ---- Phase 3 (fused): ws = sum_g weights[g] * feature(anchor_g) ----
#pragma unroll 1
  for (int m = 0; m < 16; ++m) {
    float px = posori[m * 9 + 0], py = posori[m * 9 + 1], pz = posori[m * 9 + 2];
    float a1x = posori[m * 9 + 3], a1y = posori[m * 9 + 4], a1z = posori[m * 9 + 5];
    float a2x = posori[m * 9 + 6], a2y = posori[m * 9 + 7], a2z = posori[m * 9 + 8];
    // rot6d -> R rows b1,b2,b3 (Gram-Schmidt)
    float inv1 = rsqrtf(a1x * a1x + a1y * a1y + a1z * a1z);
    float b1x = a1x * inv1, b1y = a1y * inv1, b1z = a1z * inv1;
    float d = b1x * a2x + b1y * a2y + b1z * a2z;
    float c2x = a2x - d * b1x, c2y = a2y - d * b1y, c2z = a2z - d * b1z;
    float inv2 = rsqrtf(c2x * c2x + c2y * c2y + c2z * c2z);
    float b2x = c2x * inv2, b2y = c2y * inv2, b2z = c2z * inv2;
    float b3x = b1y * b2z - b1z * b2y;
    float b3y = b1z * b2x - b1x * b2z;
    float b3z = b1x * b2y - b1y * b2x;
    float acc[4] = {0.f, 0.f, 0.f, 0.f};
#pragma unroll 1
    for (int g = 0; g < NCP; ++g) {
      float cx = s_cp[g * 3 + 0], cy = s_cp[g * 3 + 1], cz = s_cp[g * 3 + 2];
      float ax = px + b1x * cx + b1y * cy + b1z * cz;   // anchor = pos + R . cp
      float ay = py + b2x * cx + b2y * cy + b2z * cz;
      float az = pz + b3x * cx + b3y * cy + b3z * cz;
      float w = wts[m * 32 + g];
      sample_plane(pxz, ax, az, w, acc, lane);
      sample_plane(pxy, ax, ay, w, acc, lane);
      sample_plane(pyz, ay, az, w, acc, lane);
    }
#pragma unroll
    for (int j = 0; j < 4; ++j)
      ws_h[m * SROW + lane + (j << 5)] = (_Float16)acc[j];
  }
  asm volatile("s_wait_dscnt 0" ::: "memory");

  // ---- Phase 4: tmp = ws(16x128) @ W_v(128x128) + sumw * b_v (WMMA) ----
  {
    const int n = lane & 15, hif = lane >> 4;
    v16h aw[4];
#pragma unroll
    for (int k = 0; k < 4; ++k) aw[k] = load_a_frag(ws_h, k * 32, lane);
#pragma unroll
    for (int nt = 0; nt < 8; ++nt) {
      float bvn = s_bv[nt * 16 + n];
      v8f c;
#pragma unroll
      for (int r = 0; r < 8; ++r) c[r] = s_sumw[wave][r + 8 * hif] * bvn;
#pragma unroll
      for (int k = 0; k < 4; ++k) {
        v16h bf = load_b_frag_g(Wv, FDIM, nt * 16, k * 32, lane);
        c = __builtin_amdgcn_wmma_f32_16x16x32_f16(false, aw[k], false, bf,
                                                   (short)0, c, false, false);
      }
      // store tmp over ws tile (A fragments already in registers)
#pragma unroll
      for (int r = 0; r < 8; ++r)
        ws_h[(r + 8 * hif) * SROW + nt * 16 + n] = (_Float16)c[r];
    }
  }
  asm volatile("s_wait_dscnt 0" ::: "memory");

  // ---- Phase 5: out = tmp(16x128) @ W_o(128x128) + b_o + feature ----
  {
    const int n = lane & 15, hif = lane >> 4;
    v16h at[4];
#pragma unroll
    for (int k = 0; k < 4; ++k) at[k] = load_a_frag(ws_h, k * 32, lane);
#pragma unroll
    for (int nt = 0; nt < 8; ++nt) {
      v8f c = {};
#pragma unroll
      for (int k = 0; k < 4; ++k) {
        v16h bf = load_b_frag_g(Wo, FDIM, nt * 16, k * 32, lane);
        c = __builtin_amdgcn_wmma_f32_16x16x32_f16(false, at[k], false, bf,
                                                   (short)0, c, false, false);
      }
      float bon = s_bo[nt * 16 + n];
#pragma unroll
      for (int r = 0; r < 8; ++r) {
        int m = r + 8 * hif;
        float val = c[r] + bon + (float)feat_h[m * SROW + nt * 16 + n];
        out[(size_t)(pt0 + m) * FDIM + nt * 16 + n] = val;
      }
    }
  }
}

extern "C" void kernel_launch(void* const* d_in, const int* in_sizes, int n_in,
                              void* d_out, int out_size, void* d_ws, size_t ws_size,
                              hipStream_t stream) {
  (void)in_sizes; (void)n_in; (void)out_size; (void)d_ws; (void)ws_size;
  const float* qp  = (const float*)d_in[0];   // query_pos (16,1024,9)
  const float* cxz = (const float*)d_in[1];   // (16,128,128,128)
  const float* cxy = (const float*)d_in[2];
  const float* cyz = (const float*)d_in[3];
  const float* cp  = (const float*)d_in[4];   // (25,3)
  const float* Wv  = (const float*)d_in[5];   // (128,128)
  const float* bv  = (const float*)d_in[6];   // (128,)
  const float* Ww  = (const float*)d_in[7];   // (128,25)
  const float* bw  = (const float*)d_in[8];   // (25,)
  const float* Wo  = (const float*)d_in[9];   // (128,128)
  const float* bo  = (const float*)d_in[10];  // (128,)
  float* outp = (float*)d_out;                // (16,1024,128)

  // 16384 points / (4 waves * 16 points) = 256 blocks of 128 threads
  EquiGraspSO3DeformableAttn2_3410204033231_kernel<<<256, 128, 0, stream>>>(
      qp, cxz, cxy, cyz, cp, Wv, bv, Ww, bw, Wo, bo, outp);
}
